// BitNetAttention_4887672783064
// MI455X (gfx1250) — matrix operand order repaired and checked by reading, not compiled, not run on hardware
//
#include <hip/hip_runtime.h>
#include <hip/hip_bf16.h>

typedef __attribute__((ext_vector_type(16))) _Float16 v16h;
typedef __attribute__((ext_vector_type(8)))  _Float16 v8h;
typedef __attribute__((ext_vector_type(8)))  float    v8f;

// ---------------------------------------------------------------------------
// D(16x16,f32) = A(16x32,f16) * B(32x16,f16) + C
// ---------------------------------------------------------------------------
__device__ __forceinline__ v8f wmma_f16(v16h a, v16h b, v8f c) {
    return __builtin_amdgcn_wmma_f32_16x16x32_f16(
        /*neg_a=*/false, a, /*neg_b=*/false, b,
        /*c_mod=*/(short)0, c, /*reuse_a=*/false, /*reuse_b=*/false);
}

// ---------------------------------------------------------------------------
// Per-output-row binarization: s = max(mean|W[row,:]|, 1e-8)
// bw[row][k] = sign(W[row][k])  (f16, exact)
// rowscale[row] = s * learned_scale[row]
// ---------------------------------------------------------------------------
__global__ __launch_bounds__(256) void k_prep_weight(
    const float* __restrict__ w, const float* __restrict__ scale,
    _Float16* __restrict__ bw, float* __restrict__ rowscale, int in_f)
{
    const int row = blockIdx.x;
    const float* wr = w + (size_t)row * in_f;
    float acc = 0.f;
    for (int i = threadIdx.x; i < in_f; i += 256) acc += fabsf(wr[i]);
    __shared__ float red[256];
    red[threadIdx.x] = acc;
    __syncthreads();
    for (int off = 128; off > 0; off >>= 1) {
        if (threadIdx.x < off) red[threadIdx.x] += red[threadIdx.x + off];
        __syncthreads();
    }
    const float s = fmaxf(red[0] / (float)in_f, 1e-8f);
    if (threadIdx.x == 0) rowscale[row] = s * scale[row];
    _Float16* br = bw + (size_t)row * in_f;
    for (int i = threadIdx.x; i < in_f; i += 256) {
        const float v = wr[i] / s;                       // jnp.sign semantics
        const float sg = (v > 0.f) ? 1.f : ((v < 0.f) ? -1.f : 0.f);
        br[i] = (_Float16)sg;
    }
}

// ---------------------------------------------------------------------------
// f32 -> f16 elementwise
// ---------------------------------------------------------------------------
__global__ __launch_bounds__(256) void k_cvt(
    const float* __restrict__ x, _Float16* __restrict__ y, int n)
{
    const int i = blockIdx.x * 256 + threadIdx.x;
    if (i < n) y[i] = (_Float16)x[i];
}

// ---------------------------------------------------------------------------
// transpose [R][C] f16 -> [C][R]
// ---------------------------------------------------------------------------
__global__ __launch_bounds__(256) void k_transpose(
    const _Float16* __restrict__ src, _Float16* __restrict__ dst, int R, int C)
{
    const int idx = blockIdx.x * 256 + threadIdx.x;
    if (idx >= R * C) return;
    const int r = idx / C, c = idx % C;
    dst[(size_t)c * R + r] = src[idx];
}

// ---------------------------------------------------------------------------
// Y[M][N] = clamp( (A[M][K] @ BW[N][K]^T) * rowscale[N] )
// Block tile 128(M) x 128(N): 8 waves in a 4(M) x 2(N) grid, each wave owns
// 32x64 (2 M-tiles x 4 N-tiles = 8 accumulators). Per K-step of 32:
// 8 WMMAs against 12 b128 loads (2x A reuse of B, 4x B reuse of A).
// ---------------------------------------------------------------------------
template <bool OUT_F32>
__global__ __launch_bounds__(256) void k_bitgemm(
    const _Float16* __restrict__ A, const _Float16* __restrict__ BW,
    const float* __restrict__ rs, _Float16* __restrict__ Yh,
    float* __restrict__ Yf, int M, int N, int K)
{
    const int lane   = threadIdx.x & 31;
    const int wave   = threadIdx.x >> 5;
    const int waveM  = wave & 3;
    const int waveN  = wave >> 2;
    const int lane16 = lane & 15;
    const int laneHi = lane >> 4;
    const int m0 = blockIdx.y * 128 + waveM * 32;
    const int n0 = blockIdx.x * 128 + waveN * 64;

    v8f acc[2][4] = {};

    // A fragment (this code's k order, matching B): lane = M (mod 16); halves 0..15 -> K=16*laneHi..,
    // contiguous   => two adjacent v8h loads per tile.
    const _Float16* aPtr[2];
#pragma unroll
    for (int tm = 0; tm < 2; ++tm)
        aPtr[tm] = A + (size_t)(m0 + tm * 16 + lane16) * K + 16 * laneHi;
    // B fragment (same k order as A): N = lane16, K = h + 16*laneHi => contiguous v16h.
    const _Float16* bPtr[4];
#pragma unroll
    for (int t = 0; t < 4; ++t)
        bPtr[t] = BW + (size_t)(n0 + t * 16 + lane16) * K + 16 * laneHi;

    for (int k0 = 0; k0 < K; k0 += 32) {
        __builtin_prefetch(aPtr[0] + k0 + 256, 0, 1);
        __builtin_prefetch(bPtr[0] + k0 + 256, 0, 1);
        v16h a[2];
#pragma unroll
        for (int tm = 0; tm < 2; ++tm) {
            v8h a0 = *(const v8h*)(aPtr[tm] + k0);
            v8h a1 = *(const v8h*)(aPtr[tm] + k0 + 8);
#pragma unroll
            for (int i = 0; i < 8; ++i) { a[tm][i] = a0[i]; a[tm][i + 8] = a1[i]; }
        }
#pragma unroll
        for (int t = 0; t < 4; ++t) {
            v16h b = *(const v16h*)(bPtr[t] + k0);
#pragma unroll
            for (int tm = 0; tm < 2; ++tm)
                acc[tm][t] = wmma_f16(a[tm], b, acc[tm][t]);
        }
    }

    // C layout: VGPR v holds rows v (lanes 0-15) / v+8 (lanes 16-31), col = lane16.
#pragma unroll
    for (int t = 0; t < 4; ++t) {
        const int n = n0 + t * 16 + lane16;
        const float sc = rs[n];
#pragma unroll
        for (int tm = 0; tm < 2; ++tm) {
#pragma unroll
            for (int v = 0; v < 8; ++v) {
                float y = acc[tm][t][v] * sc;
                y = (y != y) ? 0.f : fminf(fmaxf(y, -1e6f), 1e6f);  // nan_to_num
                const int m = m0 + tm * 16 + v + 8 * laneHi;
                if constexpr (OUT_F32) Yf[(size_t)m * N + n] = y;
                else                   Yh[(size_t)m * N + n] = (_Float16)y;
            }
        }
    }
}

// ---------------------------------------------------------------------------
// Flash attention, GQA 32 heads / 8 kv-heads, d=128, S=2048.
// One wave per (head, 16-query tile); 4 waves per block.
// Q: [S][4096] f16 (head-major cols), K: [S][1024] f16, Vt: [1024][S] f16.
// ---------------------------------------------------------------------------
__global__ __launch_bounds__(128) void k_attn(
    const _Float16* __restrict__ Q, const _Float16* __restrict__ Km,
    const _Float16* __restrict__ Vt, _Float16* __restrict__ Ctx, int S)
{
    __shared__ _Float16 pbuf[4][16 * 32];   // per-wave P staging (C-layout -> A-layout)

    const int lane   = threadIdx.x & 31;
    const int wave   = threadIdx.x >> 5;
    const int lane16 = lane & 15;
    const int laneHi = lane >> 4;
    const int head   = blockIdx.x;
    const int kvh    = head >> 2;            // N_REP = 4
    const int q0     = (blockIdx.y * 4 + wave) * 16;

    // Q A-fragments for the 4 K-chunks of head_dim=128 (kept resident).
    const _Float16* qPtr = Q + (size_t)(q0 + lane16) * 4096 + head * 128 + 16 * laneHi;
    v16h qa[4];
#pragma unroll
    for (int c = 0; c < 4; ++c) {
        v8h x0 = *(const v8h*)(qPtr + c * 32);
        v8h x1 = *(const v8h*)(qPtr + c * 32 + 8);
#pragma unroll
        for (int i = 0; i < 8; ++i) { qa[c][i] = x0[i]; qa[c][i + 8] = x1[i]; }
    }

    const _Float16* kPtr = Km + kvh * 128 + 16 * laneHi;                 // + key*1024 + c*32
    const _Float16* vPtr = Vt + (size_t)(kvh * 128) * S + 16 * laneHi;   // + n*S + kb

    v8f   o[8] = {};
    float mrun[8], lrun[8];
#pragma unroll
    for (int v = 0; v < 8; ++v) { mrun[v] = -3.0e38f; lrun[v] = 0.f; }
    const float qscale = 0.08838834764831845f;   // 1/sqrt(128)

    _Float16* pb = &pbuf[wave][0];

    for (int kb = 0; kb < S; kb += 32) {
        // S tiles: 16 queries x 32 keys  (8 WMMAs)
        v8f st[2];
#pragma unroll
        for (int nt = 0; nt < 2; ++nt) {
            v8f s = {};
            const _Float16* kp = kPtr + (size_t)(kb + nt * 16 + lane16) * 1024;
#pragma unroll
            for (int c = 0; c < 4; ++c) {
                v16h b = *(const v16h*)(kp + c * 32);
                s = wmma_f16(qa[c], b, s);
            }
            st[nt] = s;
        }

        // Online softmax update per C-layout row slot v.
#pragma unroll
        for (int v = 0; v < 8; ++v) {
            float v0 = st[0][v] * qscale, v1 = st[1][v] * qscale;
            float tmax = fmaxf(v0, v1);
#pragma unroll
            for (int off = 1; off < 16; off <<= 1)
                tmax = fmaxf(tmax, __shfl_xor(tmax, off, 32));
            const float nm = fmaxf(mrun[v], tmax);
            const float f  = __expf(mrun[v] - nm);
            const float p0 = __expf(v0 - nm);
            const float p1 = __expf(v1 - nm);
            float rsum = p0 + p1;
#pragma unroll
            for (int off = 1; off < 16; off <<= 1)
                rsum += __shfl_xor(rsum, off, 32);
            lrun[v] = lrun[v] * f + rsum;
            mrun[v] = nm;
#pragma unroll
            for (int t = 0; t < 8; ++t) o[t][v] *= f;
            const int row = v + 8 * laneHi;
            pb[row * 32 + lane16]      = (_Float16)p0;
            pb[row * 32 + 16 + lane16] = (_Float16)p1;
        }
        __syncthreads();

        // Reload P in A-fragment layout.
        v16h pa;
        {
            const _Float16* pp = pb + lane16 * 32 + 16 * laneHi;
            v8h x0 = *(const v8h*)pp;
            v8h x1 = *(const v8h*)(pp + 8);
#pragma unroll
            for (int i = 0; i < 8; ++i) { pa[i] = x0[i]; pa[i + 8] = x1[i]; }
        }

        // O += P(16x32) @ V(32x128)   (8 WMMAs, contiguous loads from Vt)
#pragma unroll
        for (int t = 0; t < 8; ++t) {
            v16h b = *(const v16h*)(vPtr + (size_t)(t * 16 + lane16) * S + kb);
            o[t] = wmma_f16(pa, b, o[t]);
        }
        __syncthreads();
    }

    // Normalize and write context (f16, head-major columns).
#pragma unroll
    for (int t = 0; t < 8; ++t) {
#pragma unroll
        for (int v = 0; v < 8; ++v) {
            const int row = q0 + v + 8 * laneHi;
            const int col = head * 128 + t * 16 + lane16;
            Ctx[(size_t)row * 4096 + col] = (_Float16)(o[t][v] / lrun[v]);
        }
    }
}

// ---------------------------------------------------------------------------
// Host launcher
// ---------------------------------------------------------------------------
extern "C" void kernel_launch(void* const* d_in, const int* in_sizes, int n_in,
                              void* d_out, int out_size, void* d_ws, size_t ws_size,
                              hipStream_t stream) {
    (void)in_sizes; (void)n_in; (void)out_size; (void)ws_size;
    const float* hs  = (const float*)d_in[0];
    const float* qw  = (const float*)d_in[1];
    const float* qs  = (const float*)d_in[2];
    const float* kw  = (const float*)d_in[3];
    const float* ks  = (const float*)d_in[4];
    const float* vw  = (const float*)d_in[5];
    const float* vsc = (const float*)d_in[6];
    const float* ow  = (const float*)d_in[7];
    const float* osc = (const float*)d_in[8];
    float* out = (float*)d_out;

    const int S = 2048, H = 4096, KV = 1024;

    char* ws = (char*)d_ws;
    size_t off = 0;
    auto alloc = [&](size_t bytes) -> void* {
        void* p = ws + off;
        off = (off + bytes + 255) & ~(size_t)255;
        return p;
    };
    _Float16* xh  = (_Float16*)alloc((size_t)S * H * 2);
    _Float16* bwq = (_Float16*)alloc((size_t)H * H * 2);
    _Float16* bwk = (_Float16*)alloc((size_t)KV * H * 2);
    _Float16* bwv = (_Float16*)alloc((size_t)KV * H * 2);
    _Float16* bwo = (_Float16*)alloc((size_t)H * H * 2);
    float* rsq = (float*)alloc((size_t)H * 4);
    float* rsk = (float*)alloc((size_t)KV * 4);
    float* rsv = (float*)alloc((size_t)KV * 4);
    float* rso = (float*)alloc((size_t)H * 4);
    _Float16* qh  = (_Float16*)alloc((size_t)S * H * 2);
    _Float16* kh  = (_Float16*)alloc((size_t)S * KV * 2);
    _Float16* vh  = (_Float16*)alloc((size_t)S * KV * 2);
    _Float16* vt  = (_Float16*)alloc((size_t)S * KV * 2);
    _Float16* ctx = (_Float16*)alloc((size_t)S * H * 2);

    // 1) activations -> f16
    {
        const int n = S * H;
        k_cvt<<<(n + 255) / 256, 256, 0, stream>>>(hs, xh, n);
    }
    // 2) binarize weights
    k_prep_weight<<<H,  256, 0, stream>>>(qw, qs,  bwq, rsq, H);
    k_prep_weight<<<KV, 256, 0, stream>>>(kw, ks,  bwk, rsk, H);
    k_prep_weight<<<KV, 256, 0, stream>>>(vw, vsc, bwv, rsv, H);
    k_prep_weight<<<H,  256, 0, stream>>>(ow, osc, bwo, rso, H);
    // 3) Q/K/V projections (WMMA), 128x128 block tiles
    k_bitgemm<false><<<dim3(H / 128,  S / 128), 256, 0, stream>>>(xh, bwq, rsq, qh, nullptr, S, H,  H);
    k_bitgemm<false><<<dim3(KV / 128, S / 128), 256, 0, stream>>>(xh, bwk, rsk, kh, nullptr, S, KV, H);
    k_bitgemm<false><<<dim3(KV / 128, S / 128), 256, 0, stream>>>(xh, bwv, rsv, vh, nullptr, S, KV, H);
    // 4) V transpose for contiguous PV B-fragments
    {
        const int n = S * KV;
        k_transpose<<<(n + 255) / 256, 256, 0, stream>>>(vh, vt, S, KV);
    }
    // 5) flash attention (WMMA)
    k_attn<<<dim3(32, S / 64), 128, 0, stream>>>(qh, kh, vt, ctx, S);
    // 6) output projection -> f32 d_out (WMMA)
    k_bitgemm<true><<<dim3(H / 128, S / 128), 256, 0, stream>>>(ctx, bwo, rso, nullptr, out, S, H, H);
}
